// Net_11639361372195
// MI455X (gfx1250) — matrix operand order repaired and checked by reading, not compiled, not run on hardware
//
#include <hip/hip_runtime.h>
#include <hip/hip_bf16.h>
#include <stdint.h>

typedef _Float16 half_t;
typedef __attribute__((ext_vector_type(16))) _Float16 v16h;
typedef __attribute__((ext_vector_type(8)))  _Float16 v8h;
typedef __attribute__((ext_vector_type(8)))  float    v8f;

#define SEQ_T   28
#define N_IN    28
#define HID     256
#define G3      768      // 3*HID
#define KC      288      // 256 (h) + 32 (x padded); row length of WcT
#define TILE_B  32
#define BLOCK   512      // 16 waves
#define AH_STRIDE 312    // halves per A-tile row (624B -> conflict-free 16-lane stride)

#define WMMA_F16(A, B, C) \
  __builtin_amdgcn_wmma_f32_16x16x32_f16(false, (A), false, (B), (short)0, (C), false, false)

// ---------------------------------------------------------------------------
// Prep: build TRANSPOSED combined f16 weight WcT[768][288]:
//   WcT[n][k] = w_hh[n][k]      k in [0,256)
//             = w_ih[n][k-256]  k in [256,284)
//             = 0               k in [284,288)
// ---------------------------------------------------------------------------
__global__ void build_wct_kernel(const float* __restrict__ w_ih,
                                 const float* __restrict__ w_hh,
                                 half_t* __restrict__ WcT) {
  int idx = blockIdx.x * 256 + threadIdx.x;
  if (idx >= G3 * KC) return;
  int n = idx / KC;
  int k = idx - n * KC;
  float v = 0.0f;
  if (k < HID)              v = w_hh[(size_t)n * HID + k];
  else if (k < HID + N_IN)  v = w_ih[(size_t)n * N_IN + (k - HID)];
  WcT[idx] = (half_t)v;
}

// ---------------------------------------------------------------------------
// Fragment loaders (wave32, ISA 16-bit WMMA layouts)
// ---------------------------------------------------------------------------
__device__ __forceinline__ v16h load_A_frag(const half_t* Ah, int mbase, int kt, int lane) {
  int m  = mbase + (lane & 15);
  int kb = kt * 32 + ((lane < 16) ? 0 : 8);
  const half_t* p = Ah + m * AH_STRIDE + kb;
  v8h lo = *(const v8h*)(p);        // K = kb..kb+7
  v8h hi = *(const v8h*)(p + 16);   // K = kb+16..kb+23
  return __builtin_shufflevector(lo, hi, 0,1,2,3,4,5,6,7,8,9,10,11,12,13,14,15);
}

__device__ __forceinline__ v16h load_B_fragG(const half_t* __restrict__ WcT,
                                             int nbase, int kt, int lane) {
  int n  = nbase + (lane & 15);
  int kb = kt * 32 + ((lane < 16) ? 0 : 8);
  const half_t* p = WcT + (size_t)n * KC + kb;
  v8h lo = *(const v8h*)(p);
  v8h hi = *(const v8h*)(p + 16);
  return __builtin_shufflevector(lo, hi, 0,1,2,3,4,5,6,7,8,9,10,11,12,13,14,15);
}

__device__ __forceinline__ float sigmoidf_fast(float x) {
  return 1.0f / (1.0f + __expf(-x));
}

// ---------------------------------------------------------------------------
// Persistent GRU: one block owns a 32-row batch tile for all 28 timesteps.
// 16 waves; wave w owns hidden-column tile ct = w (16 cols) for all 3 gates
// and BOTH row tiles (mt = 0,1). Each B fragment feeds two WMMAs.
// Live VGPRs/wave: 8 acc frags (64) + 3 resident kt=8 B frags (24) + biases.
// ---------------------------------------------------------------------------
__global__ __launch_bounds__(BLOCK) void gru_persistent_kernel(
    const float* __restrict__ x,      // [B, T, N_IN]
    const float* __restrict__ b_ih,   // [768]
    const float* __restrict__ b_hh,   // [768]
    const float* __restrict__ fc_w,   // [28, 256]
    const float* __restrict__ fc_b,   // [28]
    const half_t* __restrict__ WcT,   // [768, 288] f16, K-contiguous
    float* __restrict__ out)          // [B, 28]
{
  __shared__ __align__(16) half_t Ah[TILE_B * AH_STRIDE]; // [32][312]: cols 0..255 = h (f16), 256..287 = x_t

  const int tid  = threadIdx.x;
  const int lane = tid & 31;
  const int wave = tid >> 5;           // 0..15 == hidden column tile ct
  const int b0   = blockIdx.x * TILE_B;
  const int gcol = lane & 15;
  const int rofs = (lane >> 4) << 3;   // 0 or 8
  const int nb   = wave * 16;          // column base within each gate region

  // init: h = 0 (and clear pad)
  for (int i = tid; i < TILE_B * AH_STRIDE; i += BLOCK) Ah[i] = (half_t)0.0f;

  // per-lane biases (constant across timesteps)
  const int g = nb + gcol;             // hidden column 0..255
  const float br  = b_ih[g]           + b_hh[g];
  const float bz  = b_ih[HID + g]     + b_hh[HID + g];
  const float bin = b_ih[2 * HID + g];
  const float bhn = b_hh[2 * HID + g];

  // t-invariant x-projection weights (K-tile 8): keep register-resident
  const v16h bR8 = load_B_fragG(WcT, 0 * HID + nb, 8, lane);
  const v16h bZ8 = load_B_fragG(WcT, 1 * HID + nb, 8, lane);
  const v16h bN8 = load_B_fragG(WcT, 2 * HID + nb, 8, lane);

  __syncthreads();

  v8f accR[2], accZ[2], accNH[2], accNI[2];   // [mt]

  for (int t = 0; t < SEQ_T; ++t) {
    // ---- stage x_t tile into Ah columns 256..287 (zero-pad 284..287) ----
    {
      int r  = tid >> 4;               // 0..31
      int c0 = (tid & 15) * 2;         // 0,2,..,30
      const float* xp = x + ((size_t)(b0 + r) * SEQ_T + t) * N_IN;
      #pragma unroll
      for (int j = 0; j < 2; ++j) {
        int c = c0 + j;
        float v = (c < N_IN) ? xp[c] : 0.0f;
        Ah[r * AH_STRIDE + HID + c] = (half_t)v;
      }
    }
    __syncthreads();   // Ah complete: h cols (prev step) + x cols (this step)

    #pragma unroll
    for (int mt = 0; mt < 2; ++mt) {
      accR[mt] = {}; accZ[mt] = {}; accNH[mt] = {}; accNI[mt] = {};
    }

    // ---- K-tiles 0..7 (h-part): rolled so B loads stay in-loop (no spills) ----
    #pragma unroll 2
    for (int kt = 0; kt < 8; ++kt) {
      v16h a0 = load_A_frag(Ah, 0,  kt, lane);
      v16h a1 = load_A_frag(Ah, 16, kt, lane);
      v16h bR = load_B_fragG(WcT, 0 * HID + nb, kt, lane);
      accR[0] = WMMA_F16(a0, bR, accR[0]);
      accR[1] = WMMA_F16(a1, bR, accR[1]);
      v16h bZ = load_B_fragG(WcT, 1 * HID + nb, kt, lane);
      accZ[0] = WMMA_F16(a0, bZ, accZ[0]);
      accZ[1] = WMMA_F16(a1, bZ, accZ[1]);
      v16h bN = load_B_fragG(WcT, 2 * HID + nb, kt, lane);
      accNH[0] = WMMA_F16(a0, bN, accNH[0]);
      accNH[1] = WMMA_F16(a1, bN, accNH[1]);
    }
    // ---- K-tile 8 (x-part): fused gi for r/z, separate i_n for n-gate ----
    {
      v16h a0 = load_A_frag(Ah, 0,  8, lane);
      v16h a1 = load_A_frag(Ah, 16, 8, lane);
      accR[0] = WMMA_F16(a0, bR8, accR[0]);
      accR[1] = WMMA_F16(a1, bR8, accR[1]);
      accZ[0] = WMMA_F16(a0, bZ8, accZ[0]);
      accZ[1] = WMMA_F16(a1, bZ8, accZ[1]);
      accNI[0] = WMMA_F16(a0, bN8, accNI[0]);
      accNI[1] = WMMA_F16(a1, bN8, accNI[1]);
    }
    __syncthreads();   // all Ah reads complete before h update

    // ---- elementwise GRU update; write new h (f16) back into Ah ----
    #pragma unroll
    for (int mt = 0; mt < 2; ++mt) {
      #pragma unroll
      for (int v = 0; v < 8; ++v) {
        int row = mt * 16 + rofs + v;
        float r_  = sigmoidf_fast(accR[mt][v] + br);
        float z_  = sigmoidf_fast(accZ[mt][v] + bz);
        float hn  = accNH[mt][v] + bhn;
        float in_ = accNI[mt][v] + bin;
        float nn  = tanhf(in_ + r_ * hn);
        float hold = (float)Ah[row * AH_STRIDE + g];
        float hnew = (1.0f - z_) * nn + z_ * hold;
        Ah[row * AH_STRIDE + g] = (half_t)hnew;
      }
    }
    // next-step x staging touches disjoint LDS columns; the barrier at the top
    // of the next iteration orders h writes before the next GEMM's reads.
  }
  __syncthreads();

  // ---- final FC: out[b0+r][o] = h[r] . fc_w[o] + fc_b[o] ----
  {
    int r = tid >> 4;                  // 0..31
    for (int o = (tid & 15); o < 28; o += 16) {
      float acc = fc_b[o];
      const float* wrow = fc_w + (size_t)o * HID;
      #pragma unroll 8
      for (int j = 0; j < HID; ++j)
        acc += (float)Ah[r * AH_STRIDE + j] * wrow[j];
      out[(size_t)(b0 + r) * 28 + o] = acc;
    }
  }
}

// ---------------------------------------------------------------------------
extern "C" void kernel_launch(void* const* d_in, const int* in_sizes, int n_in,
                              void* d_out, int out_size, void* d_ws, size_t ws_size,
                              hipStream_t stream) {
  const float* x    = (const float*)d_in[0];
  const float* w_ih = (const float*)d_in[1];
  const float* w_hh = (const float*)d_in[2];
  const float* b_ih = (const float*)d_in[3];
  const float* b_hh = (const float*)d_in[4];
  const float* fc_w = (const float*)d_in[5];
  const float* fc_b = (const float*)d_in[6];
  float* out = (float*)d_out;

  half_t* WcT = (half_t*)d_ws;   // 768*288*2 = 432 KB scratch

  build_wct_kernel<<<(G3 * KC + 255) / 256, 256, 0, stream>>>(w_ih, w_hh, WcT);

  const int batch = 4096;
  gru_persistent_kernel<<<batch / TILE_B, BLOCK, 0, stream>>>(
      x, b_ih, b_hh, fc_w, fc_b, WcT, out);
}
